// CPCSmartpoolEncoder_69990787056255
// MI455X (gfx1250) — matrix operand order repaired and checked by reading, not compiled, not run on hardware
//
#include <hip/hip_runtime.h>
#include <hip/hip_bf16.h>

// ---------------------------------------------------------------------------
// CPC Smartpool encoder for MI455X (gfx1250, wave32, WMMA).
// GEMM-shaped work (conv1-3 implicit GEMM, MLP1/2, pooling einsum) runs on
// v_wmma_f32_16x16x32_bf16 with fp32 accumulation. Aligned b128 staging goes
// through gfx1250 async global->LDS (ASYNCcnt); strides are 32-bit.
// ---------------------------------------------------------------------------

typedef __attribute__((ext_vector_type(16))) __bf16 v16bf;
typedef __attribute__((ext_vector_type(8)))  float  v8f;

union Frag { v16bf v; uint4 q[2]; };
union Q8   { uint4 q; unsigned short u[8]; };

#define EPSV 1e-5f

#if __has_builtin(__builtin_amdgcn_global_load_async_to_lds_b128)
#define HAVE_ASYNC_LDS 1
#else
#define HAVE_ASYNC_LDS 0
#endif

typedef int v4i __attribute__((vector_size(16)));
typedef __attribute__((address_space(1))) v4i as1_v4i;
typedef __attribute__((address_space(3))) v4i as3_v4i;

static __device__ __forceinline__ void copy16_g2l(unsigned short* dst,
                                                  const unsigned short* src) {
#if HAVE_ASYNC_LDS
    __builtin_amdgcn_global_load_async_to_lds_b128(
        (as1_v4i*)src, (as3_v4i*)dst, 0, 0);
#else
    *(uint4*)dst = *(const uint4*)src;
#endif
}
static __device__ __forceinline__ void wait_async() {
#if HAVE_ASYNC_LDS
#if __has_builtin(__builtin_amdgcn_s_wait_asynccnt)
    __builtin_amdgcn_s_wait_asynccnt(0);
#else
    asm volatile("s_wait_asynccnt 0" ::: "memory");
#endif
#endif
}

static __device__ __forceinline__ unsigned short f2bf(float f) {
    unsigned int u = __float_as_uint(f);
    u += 0x7FFFu + ((u >> 16) & 1u);        // round-to-nearest-even
    return (unsigned short)(u >> 16);
}
static __device__ __forceinline__ float bf2f(unsigned short h) {
    return __uint_as_float(((unsigned int)h) << 16);
}
static __device__ __forceinline__ float gelu_exact(float v) {
    return 0.5f * v * (1.0f + erff(v * 0.70710678118654752440f));
}

// ---------------------------------------------------------------------------
__global__ __launch_bounds__(256) void k_f32_to_bf16(
    const float* __restrict__ in, unsigned short* __restrict__ out, int n) {
    int i = blockIdx.x * 256 + threadIdx.x;
    if (i < n) out[i] = f2bf(in[i]);
}

// ---------------------------------------------------------------------------
// conv0 (1->512, k=10, stride=5, reflect pad 3) fused with ChannelNorm0+ReLU.
// ---------------------------------------------------------------------------
__global__ __launch_bounds__(256) void k_conv0_norm(
    const float* __restrict__ x, const float* __restrict__ w,
    const float* __restrict__ bias,
    const float* __restrict__ g, const float* __restrict__ beta,
    unsigned short* __restrict__ act, int L, int T)
{
    __shared__ float xw[10];
    __shared__ float chv[512];
    __shared__ float rs[256], rq[256];
    const int t = blockIdx.x, b = blockIdx.y, tid = threadIdx.x;

    if (tid < 10) {
        int ti = t * 5 - 3 + tid;
        if (ti < 0)   ti = -ti;
        if (ti >= L)  ti = 2 * L - 2 - ti;
        xw[tid] = x[(long)b * L + ti];
    }
    __syncthreads();

    float v[2];
    #pragma unroll
    for (int j = 0; j < 2; ++j) {
        const int c = tid + j * 256;
        float a = bias[c];
        #pragma unroll
        for (int k = 0; k < 10; ++k) a += w[c * 10 + k] * xw[k];
        chv[c] = a; v[j] = a;
    }
    rs[tid] = v[0] + v[1];
    rq[tid] = v[0] * v[0] + v[1] * v[1];
    __syncthreads();
    for (int off = 128; off > 0; off >>= 1) {
        if (tid < off) { rs[tid] += rs[tid + off]; rq[tid] += rq[tid + off]; }
        __syncthreads();
    }
    const float S = rs[0], Q = rq[0];
    const float mean = S * (1.0f / 512.0f);
    const float var  = (Q - S * mean) * (1.0f / 511.0f);     // ddof=1
    const float inv  = rsqrtf(var + EPSV);
    #pragma unroll
    for (int j = 0; j < 2; ++j) {
        const int c = tid + j * 256;
        float y = (chv[c] - mean) * inv * g[c] + beta[c];
        y = fmaxf(y, 0.0f);
        act[(long)b * 512 * T + c * T + t] = f2bf(y);
    }
}

// ---------------------------------------------------------------------------
// ChannelNorm (+ReLU) over C=512, arbitrary strides; optional bf16/f32 out.
// ---------------------------------------------------------------------------
__global__ __launch_bounds__(256) void k_cnorm(
    const float* __restrict__ in, long icb, int icc, int ict,
    const float* __restrict__ g, const float* __restrict__ beta,
    unsigned short* __restrict__ obf, long obb, int obc, int obt,
    float* __restrict__ of32, long ofb, int ofc, int oft)
{
    __shared__ float rs[256], rq[256];
    const int t = blockIdx.x, b = blockIdx.y, tid = threadIdx.x;
    const float* base = in + (long)b * icb + t * ict;
    const float v0 = base[tid * icc];
    const float v1 = base[(tid + 256) * icc];
    rs[tid] = v0 + v1;
    rq[tid] = v0 * v0 + v1 * v1;
    __syncthreads();
    for (int off = 128; off > 0; off >>= 1) {
        if (tid < off) { rs[tid] += rs[tid + off]; rq[tid] += rq[tid + off]; }
        __syncthreads();
    }
    const float S = rs[0], Q = rq[0];
    const float mean = S * (1.0f / 512.0f);
    const float var  = (Q - S * mean) * (1.0f / 511.0f);
    const float inv  = rsqrtf(var + EPSV);
    float y0 = fmaxf((v0 - mean) * inv * g[tid]       + beta[tid],       0.0f);
    float y1 = fmaxf((v1 - mean) * inv * g[tid + 256] + beta[tid + 256], 0.0f);
    if (obf) {
        obf[(long)b * obb + tid * obc + t * obt]         = f2bf(y0);
        obf[(long)b * obb + (tid + 256) * obc + t * obt] = f2bf(y1);
    }
    if (of32) {
        of32[(long)b * ofb + tid * ofc + t * oft]         = y0;
        of32[(long)b * ofb + (tid + 256) * ofc + t * oft] = y1;
    }
}

// ---------------------------------------------------------------------------
// Generic bf16 WMMA GEMM:  D[m,n] = sum_k A[m,k]*B[k,n] (+bias) (+GELU)
// 256 threads = 8 wave32; tile BM=64 x BN=128 x BK=32; wave grid 2x4;
// each wave owns 32x32 = 2x2 wmma fragments.
// KSH < 0  : B from strided memory (vectorized if sbk==1 or sbn==1)
// KSH >= 0 : B is implicit im2col over reflect-padded activations, ks = 1<<KSH
// ---------------------------------------------------------------------------
#define BM 64
#define BN 128
#define BK 32

template<int KSH>
__global__ __launch_bounds__(256) void k_gemm_bf16(
    int M, int N, int K,
    const unsigned short* __restrict__ A, int sam, int sak, long sab,
    const unsigned short* __restrict__ Bm, int sbk, int sbn, long sbb,
    const unsigned short* __restrict__ ACT, int Tin, long actb,
    int cstride, int cpad,
    const float* __restrict__ bias, int bias_mode,   // 0 none, 1 per-M, 2 per-N
    int do_gelu,
    float* __restrict__ outF, int ofm, int ofn, long ofb,
    unsigned short* __restrict__ outB, int obm, int obn, long obb)
{
    __shared__ __attribute__((aligned(16))) unsigned short As[BM][BK + 8];
    __shared__ __attribute__((aligned(16))) unsigned short Bs[BN][BK + 8];

    const int tid  = threadIdx.x;
    const int lane = tid & 31;
    const int wave = tid >> 5;
    const int waveM = wave >> 2;   // 0..1
    const int waveN = wave & 3;    // 0..3
    const int b  = blockIdx.z;
    const int m0 = blockIdx.y * BM;
    const int n0 = blockIdx.x * BN;
    const int half = lane >> 4;    // 0/1
    const int lr   = lane & 15;

    const unsigned short* Ab = A + (long)b * sab;
    const bool a_fast = (sak == 1);

    v8f acc[2][2];
    #pragma unroll
    for (int i = 0; i < 2; ++i)
        #pragma unroll
        for (int j = 0; j < 2; ++j) acc[i][j] = (v8f)(0.0f);

    for (int kk = 0; kk < K; kk += BK) {
        // ---- stage A tile ----
        if (a_fast) {
            // contiguous along k: exactly one b128 per thread
            const int m  = tid >> 2;
            const int kq = (tid & 3) * 8;
            copy16_g2l(&As[m][kq], Ab + (m0 + m) * sam + (kk + kq));
        } else {
            // strided gather (mlp1 A, einsum A: both contiguous in m, small K)
            for (int i = tid; i < BM * BK; i += 256) {
                const int m = i >> 5, k = i & 31;
                As[m][k] = Ab[(m0 + m) * sam + (kk + k) * sak];
            }
        }
        // ---- stage B tile ----
        if constexpr (KSH >= 0) {
            // implicit im2col; runs of ks=1<<KSH along k are contiguous in time
            constexpr int KS = 1 << KSH;
            const unsigned short* actp = ACT + (long)b * actb;
            for (int i = tid; i < (BN * BK) >> KSH; i += 256) {
                const int n  = i >> (5 - KSH);
                const int kq = (i & ((32 >> KSH) - 1)) << KSH;
                const int ci = (kk + kq) >> KSH;
                const unsigned short* src = actp + ci * Tin;
                const int tbase = (n0 + n) * cstride - cpad;
                if (tbase >= 0 && tbase + KS <= Tin) {
                    #pragma unroll
                    for (int j = 0; j < KS; ++j) Bs[n][kq + j] = src[tbase + j];
                } else {
                    #pragma unroll
                    for (int j = 0; j < KS; ++j) {
                        int tin = tbase + j;
                        if (tin < 0)    tin = -tin;               // reflect
                        if (tin >= Tin) tin = 2 * Tin - 2 - tin;
                        Bs[n][kq + j] = src[tin];
                    }
                }
            }
        } else {
            if (sbk == 1) {
                // contiguous along k (pooling einsum): b128 copies
                const unsigned short* Bb = Bm + (long)b * sbb;
                for (int i = tid; i < (BN * BK) / 8; i += 256) {
                    const int n  = i >> 2;
                    const int kq = (i & 3) * 8;
                    copy16_g2l(&Bs[n][kq], Bb + (n0 + n) * sbn + (kk + kq));
                }
            } else {
                // contiguous along n (w1/w2): vector load, LDS scatter
                const unsigned short* Bb = Bm + (long)b * sbb;
                for (int i = tid; i < (BN * BK) / 8; i += 256) {
                    const int k  = i >> 4;
                    const int nq = (i & 15) * 8;
                    Q8 q8;
                    q8.q = *(const uint4*)(Bb + (kk + k) * sbk + (n0 + nq));
                    #pragma unroll
                    for (int j = 0; j < 8; ++j) Bs[nq + j][k] = q8.u[j];
                }
            }
        }
        wait_async();
        __syncthreads();

        // ---- fragments per ISA 16-bit A/B VGPR striping ----
        Frag af[2], bfr[2];
        #pragma unroll
        for (int im = 0; im < 2; ++im) {
            const int mrow = waveM * 32 + im * 16 + lr;
            af[im].q[0] = *(const uint4*)&As[mrow][8 * half];
            af[im].q[1] = *(const uint4*)&As[mrow][16 + 8 * half];
        }
        #pragma unroll
        for (int jn = 0; jn < 2; ++jn) {
            const int nrow = waveN * 32 + jn * 16 + lr;
            bfr[jn].q[0] = *(const uint4*)&Bs[nrow][8 * half];
            bfr[jn].q[1] = *(const uint4*)&Bs[nrow][16 + 8 * half];
        }
        #pragma unroll
        for (int im = 0; im < 2; ++im)
            #pragma unroll
            for (int jn = 0; jn < 2; ++jn)
                acc[im][jn] = __builtin_amdgcn_wmma_f32_16x16x32_bf16(
                    false, af[im].v, false, bfr[jn].v,
                    (short)0, acc[im][jn], false, false);
        __syncthreads();
    }

    // ---- epilogue: C/D layout: vgpr r -> M = r + 8*half, N = lr ----
    #pragma unroll
    for (int im = 0; im < 2; ++im) {
        #pragma unroll
        for (int jn = 0; jn < 2; ++jn) {
            const int n = n0 + waveN * 32 + jn * 16 + lr;
            #pragma unroll
            for (int r = 0; r < 8; ++r) {
                const int m = m0 + waveM * 32 + im * 16 + 8 * half + r;
                float v = acc[im][jn][r];
                if (bias_mode == 1)      v += bias[m];
                else if (bias_mode == 2) v += bias[n];
                if (do_gelu) v = gelu_exact(v);
                if (outF) outF[(long)b * ofb + m * ofm + n * ofn] = v;
                if (outB) outB[(long)b * obb + m * obm + n * obn] = f2bf(v);
            }
        }
    }
}

// ---------------------------------------------------------------------------
// MLP head: imp[b,t] = sigmoid( z2[b,t,:] . w3 + b3 ) + 1e-5
// ---------------------------------------------------------------------------
__global__ __launch_bounds__(256) void k_head(
    const unsigned short* __restrict__ z2, const float* __restrict__ w3,
    const float* __restrict__ b3, float* __restrict__ imp, int T, int D)
{
    __shared__ float rs[256];
    const int t = blockIdx.x, b = blockIdx.y, tid = threadIdx.x;
    const unsigned short* row = z2 + ((long)b * T + t) * (long)D;
    float a = 0.0f;
    for (int i = tid; i < D; i += 256) a += bf2f(row[i]) * w3[i];
    rs[tid] = a; __syncthreads();
    for (int off = 128; off > 0; off >>= 1) {
        if (tid < off) rs[tid] += rs[tid + off];
        __syncthreads();
    }
    if (tid == 0) {
        const float lg = rs[0] + b3[0];
        imp[(long)b * T + t] = 1.0f / (1.0f + expf(-lg)) + 1e-5f;
    }
}

// ---------------------------------------------------------------------------
// Per-batch normalize + inclusive cumsum (Hillis-Steele in LDS).
// ---------------------------------------------------------------------------
__global__ __launch_bounds__(512) void k_scan(
    const float* __restrict__ imp, float* __restrict__ cs, int T, float target)
{
    __shared__ float s[512];
    const int t = threadIdx.x, b = blockIdx.x;
    s[t] = imp[(long)b * T + t];
    __syncthreads();
    for (int off = 1; off < 512; off <<= 1) {
        const float v = (t >= off) ? s[t - off] : 0.0f;
        __syncthreads();
        s[t] += v;
        __syncthreads();
    }
    const float scale = target / s[511];
    cs[(long)b * T + t] = s[t] * scale;
}

// ---------------------------------------------------------------------------
// Warp weights: wmat[b,t,n] = clamp(cs[t]-n,0,1) - clamp(cs[t-1]-n,0,1)
// ---------------------------------------------------------------------------
__global__ __launch_bounds__(256) void k_wmat(
    const float* __restrict__ cs, unsigned short* __restrict__ wm)
{
    const int idx = blockIdx.x * 256 + threadIdx.x;    // B*T*Tn = 2^19
    const int n = idx & 255;
    const int t = (idx >> 8) & 511;
    const int b = idx >> 17;
    const float cst = cs[b * 512 + t];
    const float csp = (t == 0) ? 0.0f : cs[b * 512 + t - 1];
    const float fn = (float)n;
    const float w = fminf(fmaxf(cst - fn, 0.0f), 1.0f)
                  - fminf(fmaxf(csp - fn, 0.0f), 1.0f);
    wm[idx] = f2bf(w);
}

// ---------------------------------------------------------------------------
// Orchestration
// ---------------------------------------------------------------------------
extern "C" void kernel_launch(void* const* d_in, const int* in_sizes, int n_in,
                              void* d_out, int out_size, void* d_ws, size_t ws_size,
                              hipStream_t stream) {
    constexpr int C  = 512;
    constexpr int NB = 4;
    constexpr int L  = 40960;
    constexpr int T0 = 8192, T1 = 2048, T2 = 1024, T3 = 512, TN = 256;
    constexpr int DM = 2048;

    const float* x   = (const float*)d_in[0];
    const float* c0w = (const float*)d_in[1];
    const float* c0b = (const float*)d_in[2];
    const float* c1w = (const float*)d_in[3];
    const float* c1b = (const float*)d_in[4];
    const float* c2w = (const float*)d_in[5];
    const float* c2b = (const float*)d_in[6];
    const float* c3w = (const float*)d_in[7];
    const float* c3b = (const float*)d_in[8];
    const float* w1  = (const float*)d_in[9];
    const float* b1  = (const float*)d_in[10];
    const float* w2  = (const float*)d_in[11];
    const float* b2  = (const float*)d_in[12];
    const float* w3  = (const float*)d_in[13];
    const float* b3  = (const float*)d_in[14];
    const float* n0w = (const float*)d_in[15]; const float* n0b = (const float*)d_in[16];
    const float* n1w = (const float*)d_in[17]; const float* n1b = (const float*)d_in[18];
    const float* n2w = (const float*)d_in[19]; const float* n2b = (const float*)d_in[20];
    const float* n3w = (const float*)d_in[21]; const float* n3b = (const float*)d_in[22];
    const float* n4w = (const float*)d_in[23]; const float* n4b = (const float*)d_in[24];

    size_t off = 0;
    auto take = [&](size_t bytes) {
        size_t o = off; off += (bytes + 255) & ~(size_t)255; return o;
    };
    const size_t o_wbf1 = take((size_t)C * C * 8 * 2);
    const size_t o_wbf2 = take((size_t)C * C * 4 * 2);
    const size_t o_wbf3 = take((size_t)C * C * 4 * 2);
    const size_t o_w1bf = take((size_t)C * DM * 2);
    const size_t o_w2bf = take((size_t)DM * DM * 2);
    const size_t o_act0 = take((size_t)NB * C * T0 * 2);
    const size_t o_act1 = take((size_t)NB * C * T1 * 2);
    const size_t o_act2 = take((size_t)NB * C * T2 * 2);
    const size_t o_act3 = take((size_t)NB * C * T3 * 2);
    const size_t o_z1   = take((size_t)NB * T3 * DM * 2);
    const size_t o_z2   = take((size_t)NB * T3 * DM * 2);
    const size_t o_pre  = take((size_t)NB * C * T1 * 4);   // reused pre1/2/3, pooled
    const size_t o_imp  = take((size_t)NB * T3 * 4);
    const size_t o_cs   = take((size_t)NB * T3 * 4);
    const size_t o_wm   = take((size_t)NB * T3 * TN * 2);
    if (off > ws_size) return;

    char* ws = (char*)d_ws;
    unsigned short* wbf1 = (unsigned short*)(ws + o_wbf1);
    unsigned short* wbf2 = (unsigned short*)(ws + o_wbf2);
    unsigned short* wbf3 = (unsigned short*)(ws + o_wbf3);
    unsigned short* w1bf = (unsigned short*)(ws + o_w1bf);
    unsigned short* w2bf = (unsigned short*)(ws + o_w2bf);
    unsigned short* act0 = (unsigned short*)(ws + o_act0);
    unsigned short* act1 = (unsigned short*)(ws + o_act1);
    unsigned short* act2 = (unsigned short*)(ws + o_act2);
    unsigned short* act3 = (unsigned short*)(ws + o_act3);
    unsigned short* z1   = (unsigned short*)(ws + o_z1);
    unsigned short* z2   = (unsigned short*)(ws + o_z2);
    float*          pre  = (float*)(ws + o_pre);
    float*          imp  = (float*)(ws + o_imp);
    float*          cs   = (float*)(ws + o_cs);
    unsigned short* wm   = (unsigned short*)(ws + o_wm);

    const unsigned short* NUS = nullptr;
    float* NF = nullptr;
    unsigned short* NB16 = nullptr;

    auto cvt = [&](const float* src, unsigned short* dst, long n) {
        k_f32_to_bf16<<<dim3((unsigned)((n + 255) / 256)), dim3(256), 0, stream>>>(src, dst, (int)n);
    };
    cvt(c1w, wbf1, (long)C * C * 8);
    cvt(c2w, wbf2, (long)C * C * 4);
    cvt(c3w, wbf3, (long)C * C * 4);
    cvt(w1,  w1bf, (long)C * DM);
    cvt(w2,  w2bf, (long)DM * DM);

    // conv0 + norm0 + relu (fused) -> act0 bf16 [b][c][T0]
    k_conv0_norm<<<dim3(T0, NB), dim3(256), 0, stream>>>(x, c0w, c0b, n0w, n0b, act0, L, T0);

    // conv1: M=512 N=2048 K=4096, im2col(act0, k8 s4 p2) -> pre f32 [b][co][t]
    k_gemm_bf16<3><<<dim3(T1 / BN, C / BM, NB), dim3(256), 0, stream>>>(
        C, T1, C * 8,
        wbf1, C * 8, 1, 0,
        NUS, 0, 0, 0,
        act0, T0, (long)C * T0, 4, 2,
        c1b, 1, 0,
        pre, T1, 1, (long)C * T1,
        NB16, 0, 0, 0);
    k_cnorm<<<dim3(T1, NB), dim3(256), 0, stream>>>(
        pre, (long)C * T1, T1, 1, n1w, n1b,
        act1, (long)C * T1, T1, 1, NF, 0, 0, 0);

    // conv2: M=512 N=1024 K=2048, im2col(act1, k4 s2 p1)
    k_gemm_bf16<2><<<dim3(T2 / BN, C / BM, NB), dim3(256), 0, stream>>>(
        C, T2, C * 4,
        wbf2, C * 4, 1, 0,
        NUS, 0, 0, 0,
        act1, T1, (long)C * T1, 2, 1,
        c2b, 1, 0,
        pre, T2, 1, (long)C * T2,
        NB16, 0, 0, 0);
    k_cnorm<<<dim3(T2, NB), dim3(256), 0, stream>>>(
        pre, (long)C * T2, T2, 1, n2w, n2b,
        act2, (long)C * T2, T2, 1, NF, 0, 0, 0);

    // conv3: M=512 N=512 K=2048, im2col(act2, k4 s2 p1)
    k_gemm_bf16<2><<<dim3(T3 / BN, C / BM, NB), dim3(256), 0, stream>>>(
        C, T3, C * 4,
        wbf3, C * 4, 1, 0,
        NUS, 0, 0, 0,
        act2, T2, (long)C * T2, 2, 1,
        c3b, 1, 0,
        pre, T3, 1, (long)C * T3,
        NB16, 0, 0, 0);
    k_cnorm<<<dim3(T3, NB), dim3(256), 0, stream>>>(
        pre, (long)C * T3, T3, 1, n3w, n3b,
        act3, (long)C * T3, T3, 1, NF, 0, 0, 0);

    // MLP1: z1[t,n] = gelu(f @ w1 + b1); A(m=t,k=c)=act3[c*T3+t]
    k_gemm_bf16<-1><<<dim3(DM / BN, T3 / BM, NB), dim3(256), 0, stream>>>(
        T3, DM, C,
        act3, 1, T3, (long)C * T3,
        w1bf, DM, 1, 0,
        NUS, 0, 0, 0, 0,
        b1, 2, 1,
        NF, 0, 0, 0,
        z1, DM, 1, (long)T3 * DM);

    // MLP2: z2 = gelu(z1 @ w2 + b2)
    k_gemm_bf16<-1><<<dim3(DM / BN, T3 / BM, NB), dim3(256), 0, stream>>>(
        T3, DM, DM,
        z1, DM, 1, (long)T3 * DM,
        w2bf, DM, 1, 0,
        NUS, 0, 0, 0, 0,
        b2, 2, 1,
        NF, 0, 0, 0,
        z2, DM, 1, (long)T3 * DM);

    // head, normalize+cumsum, warp weights
    k_head<<<dim3(T3, NB), dim3(256), 0, stream>>>(z2, w3, b3, imp, T3, DM);
    k_scan<<<dim3(NB), dim3(512), 0, stream>>>(imp, cs, T3, (float)TN);
    k_wmat<<<dim3((NB * T3 * TN) / 256), dim3(256), 0, stream>>>(cs, wm);

    // pooling einsum: pooled[n_new,c] = sum_t wmat[t,n_new] * f[t,c]
    // A(m=n_new,k=t)=wm[t*TN+n_new]; B(k=t,n=c)=act3[c*T3+t]
    k_gemm_bf16<-1><<<dim3(C / BN, TN / BM, NB), dim3(256), 0, stream>>>(
        TN, C, T3,
        wm, 1, TN, (long)T3 * TN,
        act3, 1, T3, (long)C * T3,
        NUS, 0, 0, 0, 0,
        NF, 0, 0,
        pre, C, 1, (long)TN * C,
        NB16, 0, 0, 0);

    // final ChannelNorm + ReLU -> d_out f32 [b][c][n]
    k_cnorm<<<dim3(TN, NB), dim3(256), 0, stream>>>(
        pre, (long)TN * C, 1, C, n4w, n4b,
        NB16, 0, 0, 0,
        (float*)d_out, (long)C * TN, TN, 1);
}
